// Decoder_75935021793458
// MI455X (gfx1250) — matrix operand order repaired and checked by reading, not compiled, not run on hardware
//
#include <hip/hip_runtime.h>
#include <hip/hip_bf16.h>
#include <stdint.h>

#define B_ 256
#define S_ 64
#define N_ 64
#define E_ 128
#define H_ 128
#define CCLIP 10.0f
#define NEGV 100000000.0f

typedef __attribute__((ext_vector_type(16))) _Float16 v16h;
typedef __attribute__((ext_vector_type(8)))  _Float16 v8h;
typedef __attribute__((ext_vector_type(4)))  _Float16 v4h;
typedef __attribute__((ext_vector_type(8)))  float    v8f;

// ---------------------------------------------------------------------------
// WMMA GEMM: C(MxN) = A(MxK) @ W(KxN) + bias(N), with W stored TRANSPOSED
// (Wt is N x K row-major) so both fragments load as contiguous b128 chunks.
// A row-major f16, C row-major f32. 4 waves per block, one 16x16 tile each.
// K, N compile-time so the K loop fully unrolls into K/32 v_wmma ops.
// ---------------------------------------------------------------------------
template <int K, int N>
__global__ __launch_bounds__(128) void gemm_wmma(const _Float16* __restrict__ A,
                                                 const _Float16* __restrict__ Wt,
                                                 const float* __restrict__ bias,
                                                 float* __restrict__ C) {
  const int tiles_n = N >> 4;
  const int tile = blockIdx.x * 4 + (threadIdx.x >> 5);
  const int tn = tile % tiles_n;
  const int tm = tile / tiles_n;
  const int lane = threadIdx.x & 31;
  const int lh   = lane >> 4;   // lane half (0: lanes 0-15, 1: lanes 16-31)
  const int l15  = lane & 15;

  v8f acc = {};
  const _Float16* Arow = A + (size_t)(tm * 16 + l15) * K;   // A row (M = l15)
  const _Float16* Wrow = Wt + (size_t)(tn * 16 + l15) * K;  // W col (N = l15)
#pragma unroll
  for (int kk = 0; kk < K; kk += 32) {
    // A fragment: K = kk + {lh*8+0..7, 16+lh*8+0..7}  -> two b128 loads
    v8h a0 = *(const v8h*)(Arow + kk + lh * 8);
    v8h a1 = *(const v8h*)(Arow + kk + 16 + lh * 8);
    // B fragment: K = kk + {lh*8+0..7, 16+lh*8+0..7}  -> two b128 loads
    v8h b0 = *(const v8h*)(Wrow + kk + lh * 8);
    v8h b1 = *(const v8h*)(Wrow + kk + 16 + lh * 8);
    v16h af = __builtin_shufflevector(a0, a1, 0, 1, 2, 3, 4, 5, 6, 7,
                                      8, 9, 10, 11, 12, 13, 14, 15);
    v16h bf = __builtin_shufflevector(b0, b1, 0, 1, 2, 3, 4, 5, 6, 7,
                                      8, 9, 10, 11, 12, 13, 14, 15);
    acc = __builtin_amdgcn_wmma_f32_16x16x32_f16(false, af, false, bf,
                                                 (short)0, acc, false, false);
  }
  const float bv = bias[tn * 16 + l15];
#pragma unroll
  for (int r = 0; r < 8; ++r)
    C[(size_t)(tm * 16 + lh * 8 + r) * N + tn * 16 + l15] = acc[r] + bv;
}

// ---------------------------------------------------------------------------
// threefry2x32 (matches JAX PRNG structure) for categorical sampling
// ---------------------------------------------------------------------------
__device__ __forceinline__ void tf_round(uint32_t& x0, uint32_t& x1, int r) {
  x0 += x1;
  x1 = (x1 << r) | (x1 >> (32 - r));
  x1 ^= x0;
}
__device__ inline void threefry2x32(uint32_t k0, uint32_t k1, uint32_t c0,
                                    uint32_t c1, uint32_t& o0, uint32_t& o1) {
  const uint32_t ks2 = 0x1BD11BDAu ^ k0 ^ k1;
  uint32_t x0 = c0 + k0, x1 = c1 + k1;
  tf_round(x0, x1, 13); tf_round(x0, x1, 15); tf_round(x0, x1, 26); tf_round(x0, x1, 6);
  x0 += k1; x1 += ks2 + 1u;
  tf_round(x0, x1, 17); tf_round(x0, x1, 29); tf_round(x0, x1, 16); tf_round(x0, x1, 24);
  x0 += ks2; x1 += k0 + 2u;
  tf_round(x0, x1, 13); tf_round(x0, x1, 15); tf_round(x0, x1, 26); tf_round(x0, x1, 6);
  x0 += k0; x1 += k1 + 3u;
  tf_round(x0, x1, 17); tf_round(x0, x1, 29); tf_round(x0, x1, 16); tf_round(x0, x1, 24);
  x0 += k1; x1 += ks2 + 4u;
  tf_round(x0, x1, 13); tf_round(x0, x1, 15); tf_round(x0, x1, 26); tf_round(x0, x1, 6);
  x0 += ks2; x1 += k0 + 5u;
  o0 = x0; o1 = x1;
}

// ---------------------------------------------------------------------------
// helper conversion / init kernels
// ---------------------------------------------------------------------------
__global__ void cvt_f16_kernel(const float* __restrict__ x,
                               _Float16* __restrict__ y, int n) {
  int i = (blockIdx.x * blockDim.x + threadIdx.x) * 4;  // n % 4 == 0
  if (i < n) {
    float4 v = *(const float4*)(x + i);
    v4h o = {(_Float16)v.x, (_Float16)v.y, (_Float16)v.z, (_Float16)v.w};
    *(v4h*)(y + i) = o;
  }
}

// x: rows x cols f32  ->  y: cols x rows f16 (transpose)
__global__ void cvt_f16_T_kernel(const float* __restrict__ x,
                                 _Float16* __restrict__ y, int rows, int cols) {
  int i = blockIdx.x * blockDim.x + threadIdx.x;
  if (i < rows * cols) {
    int r = i / cols, c = i % cols;
    y[(size_t)c * rows + r] = (_Float16)x[i];
  }
}

__global__ void mean_kernel(const float* __restrict__ cc,
                            _Float16* __restrict__ ccm) {
  int i = blockIdx.x * blockDim.x + threadIdx.x;  // over B*E
  if (i >= B_ * E_) return;
  int b = i / E_, e = i % E_;
  float s = 0.f;
  for (int t = 0; t < S_; ++t) s += cc[((size_t)b * S_ + t) * E_ + e];
  ccm[i] = (_Float16)(s * (1.0f / S_));
}

__global__ void init_iv_kernel(const float* __restrict__ init_w,
                               const float* __restrict__ vw_W,
                               const float* __restrict__ vw_b,
                               float* __restrict__ ivb) {
  int e = threadIdx.x;  // E_ threads
  float s = 0.f;
  for (int k = 0; k < 2 * E_; ++k) s += init_w[k] * vw_W[k * E_ + e];
  ivb[e] = s + vw_b[e];
}

__global__ void init_query_kernel(const float* __restrict__ h_bar,
                                  const float* __restrict__ ivb,
                                  _Float16* __restrict__ query_h) {
  int i = blockIdx.x * blockDim.x + threadIdx.x;
  if (i < B_ * E_) query_h[i] = (_Float16)(h_bar[i] + ivb[i % E_]);
}

__global__ void update_query_kernel(const float* __restrict__ h_bar,
                                    const float* __restrict__ vwtmp,
                                    _Float16* __restrict__ query_h) {
  int i = blockIdx.x * blockDim.x + threadIdx.x;
  if (i < B_ * E_) query_h[i] = (_Float16)(h_bar[i] + vwtmp[i]);
}

__global__ void init_misc_kernel(const int* __restrict__ high_mask,
                                 float* __restrict__ mask,
                                 float* __restrict__ last_node,
                                 float* __restrict__ out) {
  int i = blockIdx.x * blockDim.x + threadIdx.x;  // over B*S
  if (i < B_ * S_) mask[i] = (float)high_mask[i];
  if (i < 4 * B_) out[i] = 0.f;
  if (i < 2 * B_) last_node[i] = 0.f;
}

// ---------------------------------------------------------------------------
// Per-step attention + gumbel-argmax sample + high-level stats. 1 block per b.
// ---------------------------------------------------------------------------
__global__ __launch_bounds__(64) void attn_sample_kernel(
    const float* __restrict__ qbuf,      // (B,H)
    const float* __restrict__ keys,      // (B,S,H)
    const float* __restrict__ vvec,      // (H)
    const float* __restrict__ vb,        // scalar
    float* __restrict__ mask,            // (B,S)
    const float* __restrict__ cell_ctx,  // (B,S,E) f32
    const float* __restrict__ orig,      // (B,S,N,2)
    _Float16* __restrict__ concat_h,     // (B,2E) f16: [init_h | h]
    int* __restrict__ idxbuf,            // (B)
    float* __restrict__ last_node,       // (B,2)
    float* __restrict__ out, int step) {
  const int b = blockIdx.x;
  const int s = threadIdx.x;
  __shared__ __align__(16) float qs[H_];
  __shared__ __align__(16) float vs[H_];
  __shared__ float uarr[S_];
  __shared__ float red[S_];
  __shared__ int redi[S_];
  __shared__ int sidx;

  qs[s] = qbuf[b * H_ + s];
  qs[s + 64] = qbuf[b * H_ + s + 64];
  vs[s] = vvec[s];
  vs[s + 64] = vvec[s + 64];
  __syncthreads();

  const float4* k4 = (const float4*)(keys + ((size_t)b * S_ + s) * H_);
  float acc = 0.f;
#pragma unroll 8
  for (int h = 0; h < H_ / 4; ++h) {
    float4 kv = k4[h];
    float4 qv = ((const float4*)qs)[h];
    float4 vv = ((const float4*)vs)[h];
    acc += tanhf(qv.x + kv.x) * vv.x + tanhf(qv.y + kv.y) * vv.y +
           tanhf(qv.z + kv.z) * vv.z + tanhf(qv.w + kv.w) * vv.w;
  }
  float u = CCLIP * tanhf(acc + vb[0]) - NEGV * mask[b * S_ + s];
  uarr[s] = u;

  // logsumexp over S
  red[s] = u; __syncthreads();
  for (int off = 32; off; off >>= 1) {
    if (s < off) red[s] = fmaxf(red[s], red[s + off]);
    __syncthreads();
  }
  float m = red[0]; __syncthreads();
  red[s] = __expf(u - m); __syncthreads();
  for (int off = 32; off; off >>= 1) {
    if (s < off) red[s] += red[s + off];
    __syncthreads();
  }
  float lse = m + __logf(red[0]); __syncthreads();

  // gumbel-argmax categorical (threefry: fold_in(key(42), step) -> bits)
  uint32_t fk0, fk1;
  threefry2x32(0u, 42u, 0u, (uint32_t)step, fk0, fk1);
  const uint32_t n = (uint32_t)(B_ * S_);
  const uint32_t j = (uint32_t)(b * S_ + s);
  uint32_t r0, r1, bits;
  if (j < n / 2) { threefry2x32(fk0, fk1, j, j + n / 2, r0, r1); bits = r0; }
  else           { threefry2x32(fk0, fk1, j - n / 2, j, r0, r1); bits = r1; }
  float uni = __uint_as_float(0x3f800000u | (bits >> 9)) - 1.0f;
  uni = fmaxf(uni, 1.1754944e-38f);
  float g = -__logf(-__logf(uni));
  red[s] = u + g; redi[s] = s; __syncthreads();
  for (int off = 32; off; off >>= 1) {
    if (s < off) {
      if (red[s + off] > red[s]) { red[s] = red[s + off]; redi[s] = redi[s + off]; }
    }
    __syncthreads();
  }
  if (s == 0) sidx = (step == 0) ? 0 : redi[0];
  __syncthreads();
  const int idx = sidx;

  if (s == 0) {
    out[b] += uarr[idx] - lse;  // hlp accumulate
    const float* oc = orig + ((size_t)b * S_ + idx) * (size_t)(N_ * 2);
    float in0 = oc[0], in1 = oc[1];
    float ln0 = oc[(N_ - 1) * 2], ln1 = oc[(N_ - 1) * 2 + 1];
    float p0 = last_node[b * 2], p1 = last_node[b * 2 + 1];
    out[2 * B_ + b] += sqrtf((p0 - in0) * (p0 - in0) + (p1 - in1) * (p1 - in1));
    last_node[b * 2] = ln0;
    last_node[b * 2 + 1] = ln1;
    mask[b * S_ + idx] = fmaxf(mask[b * S_ + idx], 1.0f);
    idxbuf[b] = idx;
    out[4 * B_ + (size_t)b * S_ + step] = (float)idx;  // hact
  }
  __syncthreads();

  // h = cell_context[b, idx]; write into concat buffer (init_h at step 0)
  const float* hrow = cell_ctx + ((size_t)b * S_ + idx) * E_;
  for (int e = s; e < E_; e += 64) {
    _Float16 hv = (_Float16)hrow[e];
    concat_h[(size_t)b * 2 * E_ + E_ + e] = hv;
    if (step == 0) concat_h[(size_t)b * 2 * E_ + e] = hv;
  }
}

// ---------------------------------------------------------------------------
// Per-step low-level head: gather node_context row, softmax stats, rewards.
// 1 block (128 thr) per b. node_context gathers dominate HBM traffic, so
// loads are 128-bit and a prefetch is issued for the gathered row.
// ---------------------------------------------------------------------------
__global__ __launch_bounds__(128) void low_kernel(
    const float* __restrict__ node_ctx,  // (B,S,N,E)
    const float* __restrict__ orig,      // (B,S,N,2)
    const float* __restrict__ w_low,     // (E)
    const int* __restrict__ low_mask,    // (B,S,N)
    const int* __restrict__ idxbuf,      // (B)
    float* __restrict__ out, int step) {
  const int b = blockIdx.x;
  const int t = threadIdx.x;
  const int idx = idxbuf[b];
  __shared__ __align__(16) float wl[E_];
  __shared__ float part[128];
  __shared__ float logit[N_];
  __shared__ float red[N_];
  __shared__ int redi[N_];

  wl[t] = w_low[t];
  __syncthreads();

  const float* cur = node_ctx + ((size_t)b * S_ + idx) * (size_t)(N_ * E_);
  const int nrow = t >> 1, eh = t & 1;
  const float4* r4 = (const float4*)(cur + (size_t)nrow * E_ + eh * 64);
  __builtin_prefetch(r4, 0, 1);  // global_prefetch_b8 on gathered row
  const float4* w4 = (const float4*)(wl + eh * 64);
  float p = 0.f;
#pragma unroll
  for (int e = 0; e < 16; ++e) {
    float4 a = r4[e], w = w4[e];
    p += a.x * w.x + a.y * w.y + a.z * w.z + a.w * w.w;
  }
  part[t] = p;
  __syncthreads();
  if (eh == 0) {
    float lg = part[t] + part[t + 1];
    lg -= NEGV * (float)low_mask[((size_t)b * S_ + idx) * N_ + nrow];
    logit[nrow] = lg;
  }
  __syncthreads();

  // max
  if (t < N_) red[t] = logit[t];
  __syncthreads();
  for (int off = 32; off; off >>= 1) {
    if (t < off) red[t] = fmaxf(red[t], red[t + off]);
    __syncthreads();
  }
  float m = red[0]; __syncthreads();
  // lse
  if (t < N_) red[t] = __expf(logit[t] - m);
  __syncthreads();
  for (int off = 32; off; off >>= 1) {
    if (t < off) red[t] += red[t + off];
    __syncthreads();
  }
  float lse = m + __logf(red[0]); __syncthreads();
  // sum of logits
  if (t < N_) red[t] = logit[t];
  __syncthreads();
  for (int off = 32; off; off >>= 1) {
    if (t < off) red[t] += red[t + off];
    __syncthreads();
  }
  float sumlog = red[0]; __syncthreads();
  // argmax (first-max semantics)
  if (t < N_) { red[t] = logit[t]; redi[t] = t; }
  __syncthreads();
  for (int off = 32; off; off >>= 1) {
    if (t < off) {
      if (red[t + off] > red[t]) { red[t] = red[t + off]; redi[t] = redi[t + off]; }
    }
    __syncthreads();
  }
  int amax = redi[0];
  __syncthreads();
  // low reward: sum of segment lengths along N
  const float* oc = orig + ((size_t)b * S_ + idx) * (size_t)(N_ * 2);
  float term = 0.f;
  if (t < N_ - 1) {
    float dx = oc[2 * (t + 1)] - oc[2 * t];
    float dy = oc[2 * (t + 1) + 1] - oc[2 * t + 1];
    term = sqrtf(dx * dx + dy * dy);
  }
  if (t < N_) red[t] = term;
  __syncthreads();
  for (int off = 32; off; off >>= 1) {
    if (t < off) red[t] += red[t + off];
    __syncthreads();
  }
  if (t == 0) {
    out[B_ + b] += sumlog - (float)N_ * lse;                       // llp
    out[3 * B_ + b] += red[0];                                     // lr
    out[4 * B_ + B_ * S_ + (size_t)b * S_ + step] = (float)amax;   // lact
  }
}

// ---------------------------------------------------------------------------
extern "C" void kernel_launch(void* const* d_in, const int* in_sizes, int n_in,
                              void* d_out, int out_size, void* d_ws,
                              size_t ws_size, hipStream_t stream) {
  const float* node_ctx = (const float*)d_in[0];
  const float* orig = (const float*)d_in[1];
  const float* cc = (const float*)d_in[2];
  const int* high_mask = (const int*)d_in[3];
  const int* low_mask = (const int*)d_in[4];
  const float* init_w = (const float*)d_in[5];
  const float* h_W = (const float*)d_in[6];
  const float* h_b = (const float*)d_in[7];
  const float* vw_W = (const float*)d_in[8];
  const float* vw_b = (const float*)d_in[9];
  const float* Wq = (const float*)d_in[10];
  const float* bq = (const float*)d_in[11];
  const float* Wk = (const float*)d_in[12];
  const float* bk = (const float*)d_in[13];
  const float* vvec = (const float*)d_in[14];
  const float* vb = (const float*)d_in[15];
  const float* w_low = (const float*)d_in[16];
  float* out = (float*)d_out;
  (void)in_sizes; (void)n_in; (void)out_size; (void)ws_size;

  char* p = (char*)d_ws;
  auto alloc = [&](size_t bytes) -> char* {
    char* r = p;
    p += (bytes + 255) & ~(size_t)255;
    return r;
  };
  _Float16* cc_h = (_Float16*)alloc((size_t)B_ * S_ * E_ * 2);
  _Float16* hW_t = (_Float16*)alloc((size_t)E_ * E_ * 2);       // transposed
  _Float16* Wk_t = (_Float16*)alloc((size_t)E_ * H_ * 2);       // transposed
  _Float16* Wq_t = (_Float16*)alloc((size_t)E_ * H_ * 2);       // transposed
  _Float16* vwW_t = (_Float16*)alloc((size_t)2 * E_ * E_ * 2);  // transposed
  _Float16* ccm_h = (_Float16*)alloc((size_t)B_ * E_ * 2);
  _Float16* query_h = (_Float16*)alloc((size_t)B_ * E_ * 2);
  _Float16* concat_h = (_Float16*)alloc((size_t)B_ * 2 * E_ * 2);
  float* keys = (float*)alloc((size_t)B_ * S_ * H_ * 4);
  float* h_bar = (float*)alloc((size_t)B_ * E_ * 4);
  float* ivb = (float*)alloc((size_t)E_ * 4);
  float* qbuf = (float*)alloc((size_t)B_ * H_ * 4);
  float* vwtmp = (float*)alloc((size_t)B_ * E_ * 4);
  float* maskb = (float*)alloc((size_t)B_ * S_ * 4);
  float* lastn = (float*)alloc((size_t)B_ * 2 * 4);
  int* idxbuf = (int*)alloc((size_t)B_ * 4);

  const int T = 256;
  cvt_f16_kernel<<<(B_ * S_ * E_ / 4 + T - 1) / T, T, 0, stream>>>(cc, cc_h, B_ * S_ * E_);
  cvt_f16_T_kernel<<<(E_ * E_ + T - 1) / T, T, 0, stream>>>(h_W, hW_t, E_, E_);
  cvt_f16_T_kernel<<<(E_ * H_ + T - 1) / T, T, 0, stream>>>(Wk, Wk_t, E_, H_);
  cvt_f16_T_kernel<<<(E_ * H_ + T - 1) / T, T, 0, stream>>>(Wq, Wq_t, E_, H_);
  cvt_f16_T_kernel<<<(2 * E_ * E_ + T - 1) / T, T, 0, stream>>>(vw_W, vwW_t, 2 * E_, E_);
  mean_kernel<<<(B_ * E_ + T - 1) / T, T, 0, stream>>>(cc, ccm_h);

  // h_bar = mean_s(cc) @ h_W + h_b        (M=256, K=128, N=128), 128 tiles
  gemm_wmma<128, 128><<<(B_ / 16) * (E_ / 16) / 4, 128, 0, stream>>>(ccm_h, hW_t, h_b, h_bar);
  // keys = cc @ Wk + bk                   (M=16384, K=128, N=128), 8192 tiles
  gemm_wmma<128, 128><<<((B_ * S_) / 16) * (H_ / 16) / 4, 128, 0, stream>>>(cc_h, Wk_t, bk, keys);

  init_iv_kernel<<<1, E_, 0, stream>>>(init_w, vw_W, vw_b, ivb);
  init_query_kernel<<<(B_ * E_ + T - 1) / T, T, 0, stream>>>(h_bar, ivb, query_h);
  init_misc_kernel<<<(B_ * S_ + T - 1) / T, T, 0, stream>>>(high_mask, maskb, lastn, out);

  for (int i = 0; i < S_; ++i) {
    // q = query @ Wq + bq                 (M=256, K=128, N=128)
    gemm_wmma<128, 128><<<(B_ / 16) * (H_ / 16) / 4, 128, 0, stream>>>(query_h, Wq_t, bq, qbuf);
    attn_sample_kernel<<<B_, 64, 0, stream>>>(qbuf, keys, vvec, vb, maskb, cc, orig,
                                              concat_h, idxbuf, lastn, out, i);
    low_kernel<<<B_, 128, 0, stream>>>(node_ctx, orig, w_low, low_mask, idxbuf, out, i);
    // vwtmp = [init_h|h] @ vw_W + vw_b    (M=256, K=256, N=128)
    gemm_wmma<256, 128><<<(B_ / 16) * (E_ / 16) / 4, 128, 0, stream>>>(concat_h, vwW_t, vw_b, vwtmp);
    update_query_kernel<<<(B_ * E_ + T - 1) / T, T, 0, stream>>>(h_bar, vwtmp, query_h);
  }
}
